// STConv_29712583754085
// MI455X (gfx1250) — matrix operand order repaired and checked by reading, not compiled, not run on hardware
//
#include <hip/hip_runtime.h>
#include <hip/hip_bf16.h>
#include <stdint.h>

// ---------------------------------------------------------------------------
// STConv on MI455X (gfx1250): f16-input / f32-accumulate WMMA GEMMs.
// wave32, 8 waves / 256-thread block, 64x128 C-tile per block, K-step 32,
// double-buffered LDS software pipeline (global loads for tile k+1 overlap
// the WMMAs of tile k), 4 live B fragments so WMMAs issue back-to-back.
// ---------------------------------------------------------------------------

typedef __attribute__((ext_vector_type(16))) _Float16 v16h;
typedef __attribute__((ext_vector_type(8)))  float    v8f;

#define Bn 2
#define Sn 256
#define Nn 256
#define En 8192

// epilogue kinds
#define EPI_PLAIN    0   // C = acc
#define EPI_BIASRELU 1   // C = relu(acc + rowBias[i])
#define EPI_ACC      2   // C = C + acc (+ colBias[j] if colBias)
#define EPI_AXPBY    3   // C = alpha*acc + beta*aux[idx]

// ---------------------------------------------------------------------------
// Generic batched GEMM:  C[i,j] = sum_k A[i,k] * B(k,j)
//   BMODE 0: Bm row-major [Ncols][K]  (B(k,j) = Bm[j*ldb + k])
//   BMODE 1: Bm row-major [K][Ncols]  (B(k,j) = Bm[k*ldb + j], transposed in LDS)
// Block tile: 64 (M) x 128 (N), K-step 32. M%64==0, Ncols%128==0, K%32==0.
// ---------------------------------------------------------------------------
template <int BMODE, int EPI>
__global__ __launch_bounds__(256) void gemm_wmma_f16(
    const float* __restrict__ A,  long long strideA, int lda,
    const float* __restrict__ Bm, long long strideB, int ldb,
    float* __restrict__ C,        long long strideC, int ldc,
    const float* __restrict__ aux, long long strideAux,
    const float* __restrict__ rowBias, const float* __restrict__ colBias,
    int K, float alpha, float beta)
{
    const int j0 = blockIdx.x * 128;
    const int i0 = blockIdx.y * 64;
    const long long z = blockIdx.z;

    A  += z * strideA;
    Bm += z * strideB;
    C  += z * strideC;
    const float* auxp = aux ? aux + z * strideAux : nullptr;

    // double-buffered LDS tiles, 40-half row stride (80B rows: 16B aligned)
    __shared__ _Float16 lA[2][64 * 40];     // [m][k]
    __shared__ _Float16 lB[2][128 * 40];    // [n][k]

    const int tid  = threadIdx.x;
    const int wave = tid >> 5;
    const int lane = tid & 31;
    const int half = lane >> 4;             // 0: lanes 0-15, 1: lanes 16-31
    const int l16  = lane & 15;

    const int tr     = wave & 3;            // WMMA tile row (0..3)
    const int tcBase = (wave >> 2) * 4;     // tile cols tcBase .. tcBase+3

    v8f acc0 = {0.f,0.f,0.f,0.f,0.f,0.f,0.f,0.f};
    v8f acc1 = acc0, acc2 = acc0, acc3 = acc0;

    // per-thread staging coordinates
    const int ar  = tid >> 2;               // A: row 0..63
    const int ac0 = (tid & 3) * 8;          // A: 8 k's
    const int bn_ = tid >> 1;               // B mode0: row 0..127
    const int bc0 = (tid & 1) * 16;         // B mode0: 16 k's
    const int bk2 = (tid >> 4) * 2;         // B mode1: k-row pair 0,2..30
    const int bn0 = (tid & 15) * 8;         // B mode1: 8 n's

    // ---- register staging (pipeline stage 0) ----
    float ra[8];
    float rb[16];
    {
        const float* g = A + (long long)(i0 + ar) * lda + ac0;
        #pragma unroll
        for (int e = 0; e < 8; ++e) ra[e] = g[e];
        if (BMODE == 0) {
            const float* gb = Bm + (long long)(j0 + bn_) * ldb + bc0;
            #pragma unroll
            for (int e = 0; e < 16; ++e) rb[e] = gb[e];
        } else {
            const float* g0 = Bm + (long long)bk2 * ldb + j0 + bn0;
            const float* g1 = g0 + ldb;
            #pragma unroll
            for (int e = 0; e < 8; ++e) { rb[e] = g0[e]; rb[8 + e] = g1[e]; }
        }
    }

    int p = 0;
    for (int k0 = 0; k0 < K; k0 += 32, p ^= 1) {
        // ---- commit staged registers to LDS buffer p (fp32 -> f16) ----
        {
            _Float16* s = &lA[p][ar * 40 + ac0];
            #pragma unroll
            for (int e = 0; e < 8; ++e) s[e] = (_Float16)ra[e];
        }
        if (BMODE == 0) {
            _Float16* s = &lB[p][bn_ * 40 + bc0];
            #pragma unroll
            for (int e = 0; e < 16; ++e) s[e] = (_Float16)rb[e];
        } else {
            // pack (k, k+1) pairs -> one b32 store per n
            #pragma unroll
            for (int e = 0; e < 8; ++e) {
                union { _Float16 h[2]; uint32_t u; } pk;
                pk.h[0] = (_Float16)rb[e];
                pk.h[1] = (_Float16)rb[8 + e];
                *(uint32_t*)&lB[p][(bn0 + e) * 40 + bk2] = pk.u;
            }
        }
        __syncthreads();

        // ---- issue global loads for next K panel (consumed next iter) ----
        if (k0 + 32 < K) {
            const int kn = k0 + 32;
            const float* g = A + (long long)(i0 + ar) * lda + kn + ac0;
            #pragma unroll
            for (int e = 0; e < 8; ++e) ra[e] = g[e];
            if (BMODE == 0) {
                const float* gb = Bm + (long long)(j0 + bn_) * ldb + kn + bc0;
                #pragma unroll
                for (int e = 0; e < 16; ++e) rb[e] = gb[e];
            } else {
                const float* g0 = Bm + (long long)(kn + bk2) * ldb + j0 + bn0;
                const float* g1 = g0 + ldb;
                #pragma unroll
                for (int e = 0; e < 8; ++e) { rb[e] = g0[e]; rb[8 + e] = g1[e]; }
            }
        }

        // ---- A fragment (contiguous k = 16*half + e, same k order as B) ----
        v16h af;
        {
            const _Float16* ap = &lA[p][(tr * 16 + l16) * 40 + half * 16];
            #pragma unroll
            for (int e = 0; e < 8; ++e) { af[e] = ap[e]; af[8 + e] = ap[8 + e]; }
        }
        // ---- 4 live B fragments, then 4 back-to-back WMMAs ----
        v16h bf0, bf1, bf2, bf3;
        {
            const _Float16* b0 = &lB[p][((tcBase + 0) * 16 + l16) * 40 + half * 16];
            const _Float16* b1 = &lB[p][((tcBase + 1) * 16 + l16) * 40 + half * 16];
            const _Float16* b2 = &lB[p][((tcBase + 2) * 16 + l16) * 40 + half * 16];
            const _Float16* b3 = &lB[p][((tcBase + 3) * 16 + l16) * 40 + half * 16];
            #pragma unroll
            for (int e = 0; e < 16; ++e) {
                bf0[e] = b0[e]; bf1[e] = b1[e]; bf2[e] = b2[e]; bf3[e] = b3[e];
            }
        }
        acc0 = __builtin_amdgcn_wmma_f32_16x16x32_f16(false, af, false, bf0,
                                                      (short)0, acc0, false, false);
        acc1 = __builtin_amdgcn_wmma_f32_16x16x32_f16(false, af, false, bf1,
                                                      (short)0, acc1, false, false);
        acc2 = __builtin_amdgcn_wmma_f32_16x16x32_f16(false, af, false, bf2,
                                                      (short)0, acc2, false, false);
        acc3 = __builtin_amdgcn_wmma_f32_16x16x32_f16(false, af, false, bf3,
                                                      (short)0, acc3, false, false);
    }

    // ---- epilogue (C/D layout: VGPR r -> M = tr*16 + half*8 + r, N = l16) ----
    v8f accs[4] = {acc0, acc1, acc2, acc3};
    #pragma unroll
    for (int t = 0; t < 4; ++t) {
        const int j = j0 + (tcBase + t) * 16 + l16;
        const float cb = (EPI == EPI_ACC && colBias) ? colBias[j] : 0.f;
        #pragma unroll
        for (int r = 0; r < 8; ++r) {
            const int i = i0 + tr * 16 + half * 8 + r;
            const long long idx = (long long)i * ldc + j;
            float v = accs[t][r];
            if (EPI == EPI_BIASRELU) {
                v += rowBias[i];
                v = v > 0.f ? v : 0.f;
            } else if (EPI == EPI_ACC) {
                v += auxp[idx] + cb;
            } else if (EPI == EPI_AXPBY) {
                v = alpha * v + beta * auxp[idx];
            }
            C[idx] = v;
        }
    }
}

// ---------------------------------------------------------------------------
// Lhat build (PyG ChebConv __norm__, sym, lambda_max=2)
// ---------------------------------------------------------------------------
__global__ void lhat_zero_kernel(float* Lhat, float* deg) {
    const int i = blockIdx.x * 256 + threadIdx.x;
    if (i < Sn * Sn) Lhat[i] = 0.f;
    if (i < Sn)      deg[i]  = 0.f;
}

__global__ void lhat_deg_kernel(const int* __restrict__ ei,
                                const float* __restrict__ ew, float* deg) {
    const int e = blockIdx.x * 256 + threadIdx.x;
    if (e >= En) return;
    const int r = ei[e], c = ei[En + e];
    const float w = (r == c) ? 0.f : ew[e];
    atomicAdd(&deg[r], w);
}

__global__ void lhat_scatter_kernel(const int* __restrict__ ei,
                                    const float* __restrict__ ew,
                                    const float* __restrict__ deg,
                                    float* Lhat) {
    const int e = blockIdx.x * 256 + threadIdx.x;
    if (e >= En) return;
    const int r = ei[e], c = ei[En + e];
    const float w  = (r == c) ? 0.f : ew[e];
    const float dr = deg[r], dc = deg[c];
    const float ir = dr > 0.f ? rsqrtf(dr) : 0.f;
    const float ic = dc > 0.f ? rsqrtf(dc) : 0.f;
    atomicAdd(&Lhat[r * Sn + c], -(ir * w * ic));
}

// ---------------------------------------------------------------------------
// BatchNorm stats: channel o = axis-2 of (B,S,O,M); stats over (B,S,M)
// out flat index = z*65536 + o*256 + m,  z in [0, B*S)
// ---------------------------------------------------------------------------
__global__ __launch_bounds__(256) void bn_stats_kernel(
    const float* __restrict__ out, float* __restrict__ mu, float* __restrict__ rstd)
{
    const int o = blockIdx.x;
    const int t = threadIdx.x;
    float s = 0.f, sq = 0.f;
    for (int z = 0; z < Bn * Sn; ++z) {
        const float v = out[(long long)z * (Nn * Nn) + o * Nn + t];
        s += v; sq += v * v;
    }
    __shared__ float rs[256], rq[256];
    rs[t] = s; rq[t] = sq;
    __syncthreads();
    for (int off = 128; off > 0; off >>= 1) {
        if (t < off) { rs[t] += rs[t + off]; rq[t] += rq[t + off]; }
        __syncthreads();
    }
    if (t == 0) {
        const float cnt = (float)(Bn * Sn * Nn);
        const float m   = rs[0] / cnt;
        const float var = rq[0] / cnt - m * m;
        mu[o]   = m;
        rstd[o] = rsqrtf(var + 1e-5f);
    }
}

// ---------------------------------------------------------------------------
// y = relu( (out - mu[o]) * rstd[o] * gamma[o] + beta[o] + X )  (in place)
// float4-vectorized: 4 consecutive m share one channel o.
// ---------------------------------------------------------------------------
__global__ void bn_res_relu_kernel(float4* __restrict__ out,
                                   const float4* __restrict__ X,
                                   const float* __restrict__ mu,
                                   const float* __restrict__ rstd,
                                   const float* __restrict__ gamma,
                                   const float* __restrict__ beta,
                                   long long n4)
{
    const long long q = (long long)blockIdx.x * blockDim.x + threadIdx.x;
    if (q >= n4) return;
    const int o = (int)(((q * 4) >> 8) & 255);
    const float sc = rstd[o] * gamma[o];
    const float sh = beta[o] - mu[o] * sc;
    const float4 a = out[q];
    const float4 x = X[q];
    float4 y;
    y.x = a.x * sc + sh + x.x;
    y.y = a.y * sc + sh + x.y;
    y.z = a.z * sc + sh + x.z;
    y.w = a.w * sc + sh + x.w;
    y.x = y.x > 0.f ? y.x : 0.f;
    y.y = y.y > 0.f ? y.y : 0.f;
    y.z = y.z > 0.f ? y.z : 0.f;
    y.w = y.w > 0.f ? y.w : 0.f;
    out[q] = y;
}

// ---------------------------------------------------------------------------
// Orchestration
// ---------------------------------------------------------------------------
extern "C" void kernel_launch(void* const* d_in, const int* in_sizes, int n_in,
                              void* d_out, int out_size, void* d_ws, size_t ws_size,
                              hipStream_t stream) {
    (void)in_sizes; (void)n_in; (void)out_size; (void)ws_size;

    const float* X        = (const float*)d_in[0];
    const int*   eidx     = (const int*)  d_in[1];
    const float* ew       = (const float*)d_in[2];
    const float* tconv_w  = (const float*)d_in[3];
    const float* tconv_b  = (const float*)d_in[4];
    const float* cheb_w   = (const float*)d_in[5];   // (3, 256, 256)
    const float* cheb_b   = (const float*)d_in[6];
    const float* bn_gamma = (const float*)d_in[7];
    const float* bn_beta  = (const float*)d_in[8];
    float* out = (float*)d_out;                      // (B,S,O,M) accumulator + result

    // workspace layout
    const long long SLICE   = (long long)Nn * Nn;              // 65536
    const long long TENSOR  = (long long)Bn * Sn * SLICE;      // 33,554,432 floats
    char* ws = (char*)d_ws;
    float* T    = (float*)(ws);                                 // 134 MB (Tx0, later Tx2)
    float* Tx1  = (float*)(ws + TENSOR * 4);                    // 134 MB
    float* Lhat = (float*)(ws + 2 * TENSOR * 4);                // 256 KB
    float* deg  = (float*)(ws + 2 * TENSOR * 4 + Sn * Sn * 4);  // 1 KB
    float* mu   = deg + 256;
    float* rstd = mu + 256;

    const long long BSTR = (long long)Sn * SLICE;  // per-b stride of (S, O*N) view

    const dim3 gSlice(Nn / 128, Nn / 64, Bn * Sn);        // per-(b,s) 256^3 GEMMs
    const dim3 gSeq((Nn * Nn) / 128, Sn / 64, Bn);        // Lhat x (256 x 65536)

    // 1) Lhat = -D^-1/2 A D^-1/2  (self loops removed)
    lhat_zero_kernel<<<(Sn * Sn + 255) / 256, 256, 0, stream>>>(Lhat, deg);
    lhat_deg_kernel<<<(En + 255) / 256, 256, 0, stream>>>(eidx, ew, deg);
    lhat_scatter_kernel<<<(En + 255) / 256, 256, 0, stream>>>(eidx, ew, deg, Lhat);

    // 2) tconv: T[z][o][n] = relu( sum_c W[o,c] * X[z][n][c] + b[o] )
    gemm_wmma_f16<0, EPI_BIASRELU><<<gSlice, 256, 0, stream>>>(
        tconv_w, 0, Nn,   X, SLICE, Nn,   T, SLICE, Nn,
        nullptr, 0, tconv_b, nullptr, Nn, 1.f, 0.f);

    // 3) out = Tx0 . W0^T
    gemm_wmma_f16<0, EPI_PLAIN><<<gSlice, 256, 0, stream>>>(
        T, SLICE, Nn,   cheb_w + 0 * SLICE, 0, Nn,   out, SLICE, Nn,
        nullptr, 0, nullptr, nullptr, Nn, 1.f, 0.f);

    // 4) Tx1 = Lhat . Tx0   (contraction along S; B viewed as [256][65536])
    gemm_wmma_f16<1, EPI_PLAIN><<<gSeq, 256, 0, stream>>>(
        Lhat, 0, Sn,   T, BSTR, Nn * Nn,   Tx1, BSTR, Nn * Nn,
        nullptr, 0, nullptr, nullptr, Sn, 1.f, 0.f);

    // 5) out += Tx1 . W1^T
    gemm_wmma_f16<0, EPI_ACC><<<gSlice, 256, 0, stream>>>(
        Tx1, SLICE, Nn,   cheb_w + 1 * SLICE, 0, Nn,   out, SLICE, Nn,
        out, SLICE, nullptr, nullptr, Nn, 1.f, 0.f);

    // 6) Tx2 = 2 * Lhat . Tx1 - Tx0   (written over T; aux positions == dest)
    gemm_wmma_f16<1, EPI_AXPBY><<<gSeq, 256, 0, stream>>>(
        Lhat, 0, Sn,   Tx1, BSTR, Nn * Nn,   T, BSTR, Nn * Nn,
        T, BSTR, nullptr, nullptr, Sn, 2.f, -1.f);

    // 7) out += Tx2 . W2^T + cheb_b[m]
    gemm_wmma_f16<0, EPI_ACC><<<gSlice, 256, 0, stream>>>(
        T, SLICE, Nn,   cheb_w + 2 * SLICE, 0, Nn,   out, SLICE, Nn,
        out, SLICE, nullptr, cheb_b, Nn, 1.f, 0.f);

    // 8) BatchNorm stats per channel o, then fused BN + residual + relu
    bn_stats_kernel<<<Nn, 256, 0, stream>>>(out, mu, rstd);
    const long long n4 = TENSOR / 4;
    bn_res_relu_kernel<<<(unsigned)((n4 + 255) / 256), 256, 0, stream>>>(
        (float4*)out, (const float4*)X, mu, rstd, bn_gamma, bn_beta, n4);
}
